// LocalSelfAttentionUnFold_39256001085595
// MI455X (gfx1250) — matrix operand order repaired and checked by reading, not compiled, not run on hardware
//
#include <hip/hip_runtime.h>
#include <hip/hip_bf16.h>

// LocalSelfAttentionUnFold for MI455X (gfx1250), bf16 WMMA + flash softmax +
// TDM (tensor_load_to_lds) double-buffered K staging.
// B=4, S=2048, E=256, H=8, D=32, W=33 -> K = S-W+1 = 2016.
//
// scores[q,k] = sum_w q[q-16+w].k[k+w]  (q zero-padded), softmax over k (2016),
// out[q]      = sum_k attn[q,k] * Vs[k],  Vs[k] = sum_w v[k+w].

#define S_LEN 2048
#define E_DIM 256
#define H_NUM 8
#define D_DIM 32
#define W_WIN 33
#define K_LEN 2016
#define B_NUM 4

typedef __bf16 bf16;
typedef __attribute__((ext_vector_type(16))) __bf16 v16bf;
typedef __attribute__((ext_vector_type(8)))  __bf16 v8bf;
typedef __attribute__((ext_vector_type(8)))  float  v8f;
typedef __attribute__((ext_vector_type(4)))  float  v4f;
typedef __attribute__((ext_vector_type(4)))  unsigned int u32x4;
typedef __attribute__((ext_vector_type(8)))  int i32x8;
typedef __attribute__((ext_vector_type(4)))  int i32x4;

#if __has_builtin(__builtin_amdgcn_tensor_load_to_lds)
#define USE_TDM 1
#else
#define USE_TDM 0
#endif

__device__ __forceinline__ v16bf cat8(v8bf lo, v8bf hi) {
  v16bf r;
#pragma unroll
  for (int i = 0; i < 8; ++i) { r[i] = lo[i]; r[i + 8] = hi[i]; }
  return r;
}

// D = A(16x32 bf16) x B(32x16 bf16) + C, f32 accumulate.
__device__ __forceinline__ v8f wmma_bf16(v16bf a, v16bf b, v8f c) {
  return __builtin_amdgcn_wmma_f32_16x16x32_bf16(false, a, false, b, (short)0, c,
                                                 false, false);
}

// B-fragment: lane holds two 8-element K-chunks at +0 and +16 (one column's K-chunk).
__device__ __forceinline__ v16bf ld2x8(const bf16* p) {
  return cat8(*(const v8bf*)p, *(const v8bf*)(p + 16));
}
// A-fragment: lane holds two 8-element K-chunks at +0 and +16 of a 32-wide row.
__device__ __forceinline__ v16bf ld8_16(const bf16* p) {
  return cat8(*(const v8bf*)p, *(const v8bf*)(p + 16));
}

#if USE_TDM
// TDM: load a 2D bf16 tile (32 elements x 64 rows, row stride 32 elements)
// from global `gptr` into LDS byte offset `lds_byte_off`.
// 6-arg builtin form (amdgpu-toolchain / clang-23 lane).
__device__ __forceinline__ void tdm_load_k(const bf16* gptr, unsigned lds_byte_off) {
  unsigned long long ga = (unsigned long long)(uintptr_t)gptr;
  u32x4 g0 = { 1u,                                  // count=1, user descriptor
               lds_byte_off,                        // D#.lds_addr
               (unsigned)ga,                        // global_addr[31:0]
               (unsigned)(ga >> 32) | (2u << 30) }; // global_addr[56:32] | type=2
  i32x8 g1 = { (int)(1u << 16),    // wg_mask=0, data_size=1 (2 bytes)
               (int)(32u << 16),   // tensor_dim0 = 32 (bits 79:48 low part)
               (int)(64u << 16),   // tensor_dim0 hi=0, tensor_dim1 = 64
               (int)(32u << 16),   // tensor_dim1 hi=0, tile_dim0 = 32
               64,                 // tile_dim1 = 64, tile_dim2 = 0
               32,                 // tensor_dim0_stride = 32 (elements)
               0, 0 };
  i32x4 z4 = { 0, 0, 0, 0 };
  i32x8 z8 = { 0, 0, 0, 0, 0, 0, 0, 0 };
  __builtin_amdgcn_tensor_load_to_lds(g0, g1, z4, z4, z8, 0);
}
#endif

// ---------------------------------------------------------------------------
// Kernel 0: convert Wq/Wk/Wv (f32, [k][n]) -> bf16 transposed Wt[n][k].
// ---------------------------------------------------------------------------
__global__ void wt_convert_kernel(const float* __restrict__ Wq,
                                  const float* __restrict__ Wk,
                                  const float* __restrict__ Wv,
                                  bf16* __restrict__ Wt3) {
  int t = blockIdx.x * 256 + threadIdx.x;        // 3 * 256 * 256 total
  int proj = t >> 16;
  int nk = t & 0xFFFF;
  int n = nk >> 8, k = nk & 255;
  const float* W = (proj == 0) ? Wq : (proj == 1) ? Wk : Wv;
  Wt3[t] = (bf16)W[k * 256 + n];
}

// ---------------------------------------------------------------------------
// Kernel 1: QKV projections via WMMA. Y = x @ W + b, reshaped to heads.
// grid = (M/64, N/64, 3), block = 128 (4 waves, each 16 rows x 64 cols).
// ---------------------------------------------------------------------------
__global__ __launch_bounds__(128) void qkv_proj_kernel(
    const float* __restrict__ x, const bf16* __restrict__ Wt3,
    const float* __restrict__ bq, const float* __restrict__ bk,
    const float* __restrict__ bv, bf16* __restrict__ Qbf,
    bf16* __restrict__ Kbf, float* __restrict__ Vf) {
  const int proj = blockIdx.z;
  const bf16* Wt = Wt3 + proj * 65536;
  const float* bias = (proj == 0) ? bq : (proj == 1) ? bk : bv;
  const int tid = threadIdx.x;
  const int wv = tid >> 5, lane = tid & 31, lj = lane & 15, lh = lane >> 4;
  const int m0 = blockIdx.x * 64 + wv * 16;
  const int n0 = blockIdx.y * 64;

  v8f acc0 = {0, 0, 0, 0, 0, 0, 0, 0};
  v8f acc1 = acc0, acc2 = acc0, acc3 = acc0;

  const float* xrow = x + (size_t)(m0 + lj) * E_DIM;
#pragma unroll
  for (int kc = 0; kc < 8; ++kc) {
    const float* xp = xrow + kc * 32 + lh * 8;
    v4f f0 = *(const v4f*)(xp);
    v4f f1 = *(const v4f*)(xp + 4);
    v4f f2 = *(const v4f*)(xp + 16);
    v4f f3 = *(const v4f*)(xp + 20);
    v16bf a;
#pragma unroll
    for (int i = 0; i < 4; ++i) {
      a[i]      = (bf16)f0[i];
      a[i + 4]  = (bf16)f1[i];
      a[i + 8]  = (bf16)f2[i];
      a[i + 12] = (bf16)f3[i];
    }
    const bf16* wb = Wt + (size_t)(n0 + lj) * 256 + kc * 32 + lh * 8;
    acc0 = wmma_bf16(a, ld2x8(wb),            acc0);
    acc1 = wmma_bf16(a, ld2x8(wb + 16 * 256), acc1);
    acc2 = wmma_bf16(a, ld2x8(wb + 32 * 256), acc2);
    acc3 = wmma_bf16(a, ld2x8(wb + 48 * 256), acc3);
  }

  v8f accs[4] = {acc0, acc1, acc2, acc3};
#pragma unroll
  for (int f = 0; f < 4; ++f) {
    int n = n0 + f * 16 + lj;
    int h = n >> 5, d = n & 31;
    float bb = bias[n];
#pragma unroll
    for (int r = 0; r < 8; ++r) {
      int m = m0 + r + lh * 8;
      int b = m >> 11, s = m & 2047;
      size_t idx = (((size_t)(b * H_NUM + h)) * S_LEN + s) * D_DIM + d;
      float val = accs[f][r] + bb;
      if (proj == 0)      Qbf[idx] = (bf16)val;
      else if (proj == 1) Kbf[idx] = (bf16)val;
      else                Vf[idx]  = val;
    }
  }
}

// ---------------------------------------------------------------------------
// Kernel 2: windowed V sums, stored transposed: VsT[bh][d][k] (k stride S_LEN).
// ---------------------------------------------------------------------------
__global__ void vsum_kernel(const float* __restrict__ Vf, bf16* __restrict__ VsT) {
  int t = blockIdx.x * 256 + threadIdx.x;
  if (t >= B_NUM * H_NUM * D_DIM * K_LEN) return;
  int k = t % K_LEN;
  int rest = t / K_LEN;
  int d = rest & 31;
  int bh = rest >> 5;
  const float* vp = Vf + ((size_t)bh * S_LEN + k) * D_DIM + d;
  float s = 0.f;
#pragma unroll
  for (int w = 0; w < W_WIN; ++w) s += vp[(size_t)w * D_DIM];
  VsT[((size_t)bh * D_DIM + d) * S_LEN + k] = (bf16)s;
}

// ---------------------------------------------------------------------------
// Kernel 3: flash attention. grid = (B*H, S/64), block = 128 (4 waves).
// Wave wv handles 16 query rows q0 = 64*blockIdx.y + 16*wv.
// K tiles (64 rows x 32 bf16) staged into LDS by the Tensor Data Mover,
// double-buffered behind the WMMA stream.
// ---------------------------------------------------------------------------
__global__ __launch_bounds__(128) void attn_kernel(
    const bf16* __restrict__ Qbf, const bf16* __restrict__ Kbf,
    const bf16* __restrict__ VsT, float* __restrict__ out) {
  __shared__ __align__(16) bf16 Qs[112 * 32];      // rows q0blk-16 .. q0blk+95
  __shared__ __align__(16) bf16 Ks[2][64 * 32];    // double-buffered K tiles
  __shared__ __align__(16) bf16 Pb[4][16 * 32];    // per-wave P transpose buf

  const int bh = blockIdx.x;
  const int q0blk = blockIdx.y * 64;
  const int tid = threadIdx.x;
  const int wv = tid >> 5, lane = tid & 31, lj = lane & 15, lh = lane >> 4;
  const float scale = 0.17677669529663687f;        // 1/sqrt(32)

  const bf16* Kg = Kbf + (size_t)bh * S_LEN * D_DIM;  // row k -> Kg + k*32

#if USE_TDM
  // Kick off the first K tile before Q staging so the DMA overlaps it.
  if (wv == 0) tdm_load_k(Kg, (unsigned)(uintptr_t)&Ks[0][0]);
#endif

  // Stage Q rows (zero-padded outside [0,S)) into LDS.
  const bf16* Qb = Qbf + (size_t)bh * S_LEN * D_DIM;
  for (int j = tid; j < 112 * 16; j += 128) {      // one dword (2 bf16) each
    int row = j >> 4, cp = j & 15;
    int q = q0blk - 16 + row;
    unsigned val = 0u;
    if (q >= 0 && q < S_LEN) val = *(const unsigned*)(Qb + (size_t)q * 32 + cp * 2);
    *(unsigned*)(Qs + row * 32 + cp * 2) = val;
  }
  __syncthreads();

  const bf16* abase  = Qs + (16 * wv + lj) * 32 + lh * 8;   // A-frag base (w=0)
  const bf16* v0base = VsT + (size_t)bh * D_DIM * S_LEN + (size_t)lj * S_LEN + lh * 8;
  const bf16* v1base = v0base + (size_t)16 * S_LEN;
  bf16* pwb = &Pb[wv][0];
  const bf16* prb = pwb + lj * 32 + lh * 8;

  float m_r[8], l_r[8];
  v8f o0 = {0, 0, 0, 0, 0, 0, 0, 0}, o1 = o0;
#pragma unroll
  for (int r = 0; r < 8; ++r) { m_r[r] = -3.0e38f; l_r[r] = 0.f; }

#pragma unroll 1
  for (int kt = 0; kt < 63; ++kt) {
    const int k0 = kt * 32;

#if USE_TDM
    // Issue next tile (kt=62 issues a dummy tile into the dead buffer so the
    // tensorcnt wait immediate stays a constant 1; it lands inside d_ws).
    if (wv == 0) {
      tdm_load_k(Kg + (size_t)(k0 + 32) * D_DIM,
                 (unsigned)(uintptr_t)&Ks[(kt + 1) & 1][0]);
      __builtin_amdgcn_s_wait_tensorcnt(1);   // tile kt complete (in-order)
    }
    __syncthreads();                          // publish tile kt to all waves
    const bf16* kb = &Ks[kt & 1][0] + lj * 32 + lh * 8;
#else
    // Fallback: cooperative synchronous staging (single buffer).
    __syncthreads();
    for (int j = tid; j < 64 * 16; j += 128) {
      int row = j >> 4, cp = j & 15;
      *(unsigned*)(&Ks[0][0] + row * 32 + cp * 2) =
          *(const unsigned*)(Kg + (size_t)(k0 + row) * 32 + cp * 2);
    }
    __syncthreads();
    const bf16* kb = &Ks[0][0] + lj * 32 + lh * 8;
#endif

    // scores(16x32) = sum_w Q[q0-16+w:+16,:] x K[k0+{0,16}+w:+16,:]^T
    // (local LDS tile row r0 = global row k0 + r0)
    v8f s0a = {0, 0, 0, 0, 0, 0, 0, 0};
    v8f s0b = s0a, s1a = s0a, s1b = s0a;
#pragma unroll 4
    for (int w = 0; w < 32; w += 2) {
      v16bf a0 = ld8_16(abase + w * 32);
      v16bf a1 = ld8_16(abase + (w + 1) * 32);
      s0a = wmma_bf16(a0, ld2x8(kb + (w)      * 32), s0a);
      s1a = wmma_bf16(a0, ld2x8(kb + (w + 16) * 32), s1a);
      s0b = wmma_bf16(a1, ld2x8(kb + (w + 1)  * 32), s0b);
      s1b = wmma_bf16(a1, ld2x8(kb + (w + 17) * 32), s1b);
    }
    {
      v16bf a = ld8_16(abase + 32 * 32);
      s0a = wmma_bf16(a, ld2x8(kb + 32 * 32), s0a);
      s1a = wmma_bf16(a, ld2x8(kb + 48 * 32), s1a);
    }
    v8f s0 = s0a + s0b, s1 = s1a + s1b;

    // Online softmax update. C-layout: slot r, lane -> row M = r + 8*lh,
    // column = lj (s0) / 16+lj (s1); reduce across the 16 lanes of each half.
    float p0[8], p1[8];
#pragma unroll
    for (int r = 0; r < 8; ++r) {
      float x0 = s0[r] * scale, x1 = s1[r] * scale;
      float t = fmaxf(x0, x1);
      t = fmaxf(t, __shfl_xor(t, 1, 32));
      t = fmaxf(t, __shfl_xor(t, 2, 32));
      t = fmaxf(t, __shfl_xor(t, 4, 32));
      t = fmaxf(t, __shfl_xor(t, 8, 32));
      float mn = fmaxf(m_r[r], t);
      float al = __expf(m_r[r] - mn);
      m_r[r] = mn;
      float e0 = __expf(x0 - mn), e1 = __expf(x1 - mn);
      float u = e0 + e1;
      u += __shfl_xor(u, 1, 32);
      u += __shfl_xor(u, 2, 32);
      u += __shfl_xor(u, 4, 32);
      u += __shfl_xor(u, 8, 32);
      l_r[r] = l_r[r] * al + u;
      o0[r] *= al;
      o1[r] *= al;
      p0[r] = e0;
      p1[r] = e1;
    }

    // Transpose P (C-layout -> A-layout) through per-wave LDS buffer.
#pragma unroll
    for (int r = 0; r < 8; ++r) {
      int row = r + lh * 8;
      pwb[row * 32 + lj]      = (bf16)p0[r];
      pwb[row * 32 + 16 + lj] = (bf16)p1[r];
    }
    asm volatile("s_wait_dscnt 0" ::: "memory");   // same-wave LDS RAW
    v16bf pf = ld8_16(prb);

    // out(16x32) += P(16x32) x Vs(32x32)
    o0 = wmma_bf16(pf, ld2x8(v0base + k0), o0);
    o1 = wmma_bf16(pf, ld2x8(v1base + k0), o1);

    __syncthreads();   // all waves done with tile kt before its buffer reuse
  }

  // Normalize and write out[b][q][h*32+d].
  const int b_ = bh >> 3, h_ = bh & 7;
#pragma unroll
  for (int r = 0; r < 8; ++r) {
    int q = q0blk + wv * 16 + r + lh * 8;
    float inv = 1.0f / l_r[r];
    float* op = out + ((size_t)(b_ * S_LEN + q)) * E_DIM + h_ * 32;
    op[lj]      = o0[r] * inv;
    op[16 + lj] = o1[r] * inv;
  }
}

// ---------------------------------------------------------------------------
// Workspace layout (requires ~20.4 MB):
//   Wt3  bf16  3*256*256            @ 0
//   Qbf  bf16  4*8*2048*32          @ 0x060000
//   Kbf  bf16  4*8*2048*32          @ 0x460000
//   Vf   f32   4*8*2048*32          @ 0x860000
//   VsT  bf16  4*8*32*2048          @ 0x1060000
// ---------------------------------------------------------------------------
extern "C" void kernel_launch(void* const* d_in, const int* in_sizes, int n_in,
                              void* d_out, int out_size, void* d_ws, size_t ws_size,
                              hipStream_t stream) {
  (void)in_sizes; (void)n_in; (void)out_size; (void)ws_size;
  const float* x  = (const float*)d_in[0];
  const float* Wq = (const float*)d_in[1];
  const float* bq = (const float*)d_in[2];
  const float* Wk = (const float*)d_in[3];
  const float* bk = (const float*)d_in[4];
  const float* Wv = (const float*)d_in[5];
  const float* bv = (const float*)d_in[6];

  char* ws = (char*)d_ws;
  bf16*  Wt3 = (bf16*)(ws);
  bf16*  Qbf = (bf16*)(ws + 0x060000);
  bf16*  Kbf = (bf16*)(ws + 0x460000);
  float* Vf  = (float*)(ws + 0x860000);
  bf16*  VsT = (bf16*)(ws + 0x1060000);

  wt_convert_kernel<<<768, 256, 0, stream>>>(Wq, Wk, Wv, Wt3);
  qkv_proj_kernel<<<dim3(128, 4, 3), 128, 0, stream>>>(x, Wt3, bq, bk, bv,
                                                       Qbf, Kbf, Vf);
  int vtot = B_NUM * H_NUM * D_DIM * K_LEN;
  vsum_kernel<<<(vtot + 255) / 256, 256, 0, stream>>>(Vf, VsT);
  attn_kernel<<<dim3(B_NUM * H_NUM, S_LEN / 64), 128, 0, stream>>>(
      Qbf, Kbf, VsT, (float*)d_out);
}
